// EQNetwork_75419625718090
// MI455X (gfx1250) — matrix operand order repaired and checked by reading, not compiled, not run on hardware
//
#include <hip/hip_runtime.h>

typedef __attribute__((ext_vector_type(16))) _Float16 v16h;
typedef __attribute__((ext_vector_type(8)))  float    v8f;
typedef __attribute__((ext_vector_type(4)))  float    v4f;

#define KNBR   16
#define NB     10
#define HDIM   100
#define RSTEP  (5.0f/9.0f)

union AFrag { v16h v; v4f f[2]; };

__device__ __forceinline__ v8f wmma_f16(v16h a, v16h b, v8f c) {
  // D = A(16x32 f16) * B(32x16 f16) + C(16x16 f32)
  return __builtin_amdgcn_wmma_f32_16x16x32_f16(false, a, false, b, (short)0, c, false, false);
}

// ---------------------------------------------------------------------------
// Pack f32 weights [ksrc x nsrc] into per-lane WMMA B-fragment order (f16):
// dst[((t*kkn+kk)*32 + L)*16 + e] = w[kk*32 + (e<8 ? e : e+8) + 8*(L/16)][t*16 + L%16]
// (zero-padded outside [ksrc, nsrc))
// ---------------------------------------------------------------------------
__global__ void pack_w(const float* __restrict__ src, _Float16* __restrict__ dst,
                       int ksrc, int nsrc, int ntile, int kkn) {
  int total = ntile * kkn * 512;
  for (int idx = blockIdx.x * blockDim.x + threadIdx.x; idx < total;
       idx += gridDim.x * blockDim.x) {
    int e  = idx & 15;
    int L  = (idx >> 4) & 31;
    int tk = idx >> 9;
    int kk = tk % kkn;
    int t  = tk / kkn;
    int row = kk * 32 + ((e < 8) ? e : e + 8) + 8 * (L >> 4);
    int col = t * 16 + (L & 15);
    float v = (row < ksrc && col < nsrc) ? src[(size_t)row * nsrc + col] : 0.0f;
    dst[idx] = (_Float16)v;
  }
}

__global__ void prep_f0(const int* __restrict__ charges, float* __restrict__ f0, int n) {
  int i = blockIdx.x * blockDim.x + threadIdx.x;
  if (i < n) f0[i] = (float)charges[i] * (1.0f / 94.0f) - 0.5f;
}

// ---------------------------------------------------------------------------
// Fused layer kernel: 2 waves/block, 2 nodes/wave. Each node's 16 edges form
// one WMMA M-tile; each B fragment is loaded once and reused for both nodes.
//   basis -> h1 (7 WMMA tiles) -> h2 (28) -> W tiles (NT*4) -> tensor product
//   (all 32 lanes; lane owns node=lane/16, edge=lane%16) -> scatter-mean ->
//   gated nonlinearity.
// ---------------------------------------------------------------------------
template<int M0I, int M1I, int M0O, int GG, int M1O, int NOUT>
__global__ __launch_bounds__(64) void layer_kernel(
    const float* __restrict__ radii,    // [N*16*3]
    const int*   __restrict__ nbr,      // [N*16]
    const float* __restrict__ f0in,     // [N*M0I]
    const float* __restrict__ f1in,     // [N*M1I*3] (or null)
    const _Float16* __restrict__ w1p,
    const _Float16* __restrict__ w2p,
    const _Float16* __restrict__ w3p,
    const float* __restrict__ b1v,
    const float* __restrict__ b2v,
    float* __restrict__ f0out,          // [N*M0O]
    float* __restrict__ f1out)          // [N*M1O*3] (or null)
{
  constexpr int M0T  = M0O + GG;
  constexpr int NT   = NOUT / 16;
  constexpr int M1IS = (M1I > 0) ? M1I : 1;
  constexpr int WAVES = 2;
  constexpr int NPW   = 2;              // nodes (M-tiles) per wave

  __shared__ alignas(16) _Float16 h1[WAVES][NPW * 16][128];
  __shared__ alignas(16) _Float16 h2[WAVES][NPW * 16][128];
  __shared__ alignas(16) float tile[WAVES][NPW * 16][16];
  __shared__ float g0s[WAVES][NPW * 16][M0I];
  __shared__ float g1s[WAVES][NPW * 16][M1IS * 3];
  __shared__ float g1d[WAVES][NPW * 16][M1IS];
  __shared__ float sac[WAVES][NPW * 16][M0T];
  __shared__ float vac[WAVES][NPW * 16][((M1O > 0) ? M1O : 1) * 3];
  __shared__ float gate[WAVES][NPW][(GG > 0) ? GG : 1];

  const int wv   = threadIdx.x >> 5;
  const int lane = threadIdx.x & 31;
  const int g    = lane >> 4;           // half-wave group == owned node (for TP)
  const int ln   = lane & 15;           // matrix row M == edge within node
  const int n0   = (blockIdx.x * WAVES + wv) * NPW;

  __builtin_prefetch(w2p, 0, 3);
  __builtin_prefetch(w3p, 0, 3);

  // zero activation buffers (covers padded K region 112..127)
  {
    v4f zz = {};
    v4f* z1 = (v4f*)&h1[wv][0][0];
    v4f* z2 = (v4f*)&h2[wv][0][0];
#pragma unroll
    for (int i = 0; i < 16; ++i) { z1[lane + 32 * i] = zz; z2[lane + 32 * i] = zz; }
  }

  // geometry: every lane loads edge `ln` of BOTH nodes (needed for A layout)
  float rr[NPW], ux[NPW], uy[NPW], uz[NPW];
#pragma unroll
  for (int j = 0; j < NPW; ++j) {
    size_t e = ((size_t)(n0 + j) * KNBR + ln) * 3;
    float rx = radii[e + 0], ry = radii[e + 1], rz = radii[e + 2];
    float r  = sqrtf(rx * rx + ry * ry + rz * rz);
    float iv = 1.0f / (r + 1e-8f);
    rr[j] = r; ux[j] = rx * iv; uy[j] = ry * iv; uz[j] = rz * iv;
  }
  // per-lane u of the OWNED node (no dynamic register indexing)
  const float uX = g ? ux[1] : ux[0];
  const float uY = g ? uy[1] : uy[0];
  const float uZ = g ? uz[1] : uz[0];

  // A1 fragments: cosine-bump basis in the documented WMMA A layout
  v16h a1[NPW];
#pragma unroll
  for (int j = 0; j < NPW; ++j) {
#pragma unroll
    for (int e = 0; e < 16; ++e) {
      int kidx = ((e < 8) ? e : e + 8) + g * 8;     // K index of half e
      float d  = rr[j] * (1.0f / RSTEP) - (float)kidx;
      float bb = 0.0f;
      if (kidx < NB && fabsf(d) < 1.0f) { float cd = __cosf(1.57079632679f * d); bb = cd * cd; }
      a1[j][e] = (_Float16)bb;
    }
  }

  // -------- GEMM1: basis -> h1, fused bias+ReLU (B loaded once per t) -------
#pragma unroll
  for (int t = 0; t < 7; ++t) {
    v16h b = *(const v16h*)(w1p + ((size_t)t * 32 + lane) * 16);
    int col = t * 16 + ln;
    float bias = (col < HDIM) ? b1v[col] : 0.0f;
#pragma unroll
    for (int j = 0; j < NPW; ++j) {
      v8f acc = {};
      acc = wmma_f16(a1[j], b, acc);
#pragma unroll
      for (int i = 0; i < 8; ++i) {
        float v = acc[i] + bias;
        h1[wv][j * 16 + i + 8 * g][col] = (_Float16)((v > 0.0f) ? v : 0.0f);
      }
    }
  }
  __syncthreads();

  // A2 fragments from LDS (two 16B reads per k-step, documented A layout)
  AFrag a2[NPW][4];
#pragma unroll
  for (int j = 0; j < NPW; ++j) {
    const v4f* hrow = (const v4f*)&h1[wv][j * 16 + ln][0];
#pragma unroll
    for (int kk = 0; kk < 4; ++kk) { a2[j][kk].f[0] = hrow[kk * 4 + g]; a2[j][kk].f[1] = hrow[kk * 4 + g + 2]; }
  }

  // -------- GEMM2: h1 -> h2, fused bias+ReLU --------
#pragma unroll
  for (int t = 0; t < 7; ++t) {
    v8f acc[NPW];
#pragma unroll
    for (int j = 0; j < NPW; ++j) acc[j] = (v8f){};
#pragma unroll
    for (int kk = 0; kk < 4; ++kk) {
      v16h b = *(const v16h*)(w2p + ((size_t)(t * 4 + kk) * 32 + lane) * 16);
#pragma unroll
      for (int j = 0; j < NPW; ++j) acc[j] = wmma_f16(a2[j][kk].v, b, acc[j]);
    }
    int col = t * 16 + ln;
    float bias = (col < HDIM) ? b2v[col] : 0.0f;
#pragma unroll
    for (int j = 0; j < NPW; ++j) {
#pragma unroll
      for (int i = 0; i < 8; ++i) {
        float v = acc[j][i] + bias;
        h2[wv][j * 16 + i + 8 * g][col] = (_Float16)((v > 0.0f) ? v : 0.0f);
      }
    }
  }
  __syncthreads();

  AFrag a3[NPW][4];
#pragma unroll
  for (int j = 0; j < NPW; ++j) {
    const v4f* hrow = (const v4f*)&h2[wv][j * 16 + ln][0];
#pragma unroll
    for (int kk = 0; kk < 4; ++kk) { a3[j][kk].f[0] = hrow[kk * 4 + g]; a3[j][kk].f[1] = hrow[kk * 4 + g + 2]; }
  }

  // gather neighbor features (with per-layer /2) + init accumulators.
  // Every lane owns (node g, edge ln) -> all 32 lanes active.
  {
    const int rrow = g * 16 + ln;
    const int edge = (n0 + g) * KNBR + ln;
    int nidx = nbr[edge];
#pragma unroll
    for (int i = 0; i < M0I; ++i) g0s[wv][rrow][i] = 0.5f * f0in[(size_t)nidx * M0I + i];
    if constexpr (M1I > 0) {
#pragma unroll
      for (int i = 0; i < M1I; ++i) {
        float gx = 0.5f * f1in[((size_t)nidx * M1I + i) * 3 + 0];
        float gy = 0.5f * f1in[((size_t)nidx * M1I + i) * 3 + 1];
        float gz = 0.5f * f1in[((size_t)nidx * M1I + i) * 3 + 2];
        g1s[wv][rrow][i * 3 + 0] = gx;
        g1s[wv][rrow][i * 3 + 1] = gy;
        g1s[wv][rrow][i * 3 + 2] = gz;
        g1d[wv][rrow][i] = gx * uX + gy * uY + gz * uZ;
      }
    }
#pragma unroll
    for (int o = 0; o < M0T; ++o) sac[wv][rrow][o] = 0.0f;
    if constexpr (M1O > 0) {
#pragma unroll
      for (int j = 0; j < M1O * 3; ++j) vac[wv][rrow][j] = 0.0f;
    }
  }

  constexpr int O1 = M0T * M0I;
  constexpr int O2 = O1 + M0T * M1I;
  constexpr int O3 = O2 + M1O * M0I;
  constexpr int O4 = O3 + M1O * M1I;

  // -------- GEMM3 + equivariant tensor product, one 16x16 tile at a time ----
  for (int t = 0; t < NT; ++t) {
    v8f acc[NPW];
#pragma unroll
    for (int j = 0; j < NPW; ++j) acc[j] = (v8f){};
#pragma unroll
    for (int kk = 0; kk < 4; ++kk) {
      v16h b = *(const v16h*)(w3p + ((size_t)(t * 4 + kk) * 32 + lane) * 16);
#pragma unroll
      for (int j = 0; j < NPW; ++j) acc[j] = wmma_f16(a3[j][kk].v, b, acc[j]);
    }
    __syncthreads();                 // previous tile fully consumed
#pragma unroll
    for (int j = 0; j < NPW; ++j)
#pragma unroll
      for (int i = 0; i < 8; ++i) tile[wv][j * 16 + i + 8 * g][ln] = acc[j][i];
    __syncthreads();
    {
      const int rrow = g * 16 + ln;  // lane's owned (node, edge) row
#pragma unroll
      for (int c = 0; c < 16; ++c) {
        int C = t * 16 + c;
        float w = tile[wv][rrow][c];
        if (C < O1) {                                   // 0 -> 0
          int o = C / M0I, i = C % M0I;
          sac[wv][rrow][o] += w * g0s[wv][rrow][i];
        } else if (M1I > 0 && C < O2) {                 // 1 -> 0 (dot with u)
          int d = C - O1; int o = d / M1IS, i = d % M1IS;
          sac[wv][rrow][o] += w * g1d[wv][rrow][i];
        } else if (M1O > 0 && C < O3) {                 // 0 -> 1 (times u)
          int d = C - O2; int o = d / M0I, i = d % M0I;
          float m = w * g0s[wv][rrow][i];
          vac[wv][rrow][o * 3 + 0] += m * uX;
          vac[wv][rrow][o * 3 + 1] += m * uY;
          vac[wv][rrow][o * 3 + 2] += m * uZ;
        } else if (M1O > 0 && M1I > 0 && C < O4) {      // 1 -> 1, l_f = 0
          int d = C - O3; int o = d / M1IS, i = d % M1IS;
          vac[wv][rrow][o * 3 + 0] += w * g1s[wv][rrow][i * 3 + 0];
          vac[wv][rrow][o * 3 + 1] += w * g1s[wv][rrow][i * 3 + 1];
          vac[wv][rrow][o * 3 + 2] += w * g1s[wv][rrow][i * 3 + 2];
        } else if (M1O > 0 && M1I > 0) {                // 1 -> 1, l_f = 1 (cross)
          int d = C - O4; int o = d / M1IS, i = d % M1IS;
          float gx = g1s[wv][rrow][i * 3 + 0];
          float gy = g1s[wv][rrow][i * 3 + 1];
          float gz = g1s[wv][rrow][i * 3 + 2];
          vac[wv][rrow][o * 3 + 0] += w * (gy * uZ - gz * uY);
          vac[wv][rrow][o * 3 + 1] += w * (gz * uX - gx * uZ);
          vac[wv][rrow][o * 3 + 2] += w * (gx * uY - gy * uX);
        }
      }
    }
  }
  __syncthreads();

  // scatter-mean over 16 neighbors + gated nonlinearity, per node
#pragma unroll
  for (int j = 0; j < NPW; ++j) {
    if (lane < M0T) {
      float s = 0.0f;
#pragma unroll
      for (int e = 0; e < 16; ++e) s += sac[wv][j * 16 + e][lane];
      float mean = s * (1.0f / 16.0f);
      if (lane < M0O) {
        f0out[(size_t)(n0 + j) * M0O + lane] = (mean > 0.0f) ? mean : 0.0f;
      } else if constexpr (GG > 0) {
        gate[wv][j][lane - M0O] = 1.0f / (1.0f + __expf(-mean));
      }
    }
  }
  if constexpr (M1O > 0) {
    __syncthreads();
#pragma unroll
    for (int j = 0; j < NPW; ++j) {
      if (lane < M1O * 3) {
        float s = 0.0f;
#pragma unroll
        for (int e = 0; e < 16; ++e) s += vac[wv][j * 16 + e][lane];
        f1out[(size_t)(n0 + j) * M1O * 3 + lane] = s * (1.0f / 16.0f) * gate[wv][j][lane / 3];
      }
    }
  }
}

// ---------------------------------------------------------------------------
// Deterministic two-pass mean over nodes of f0 [N,32] -> out [32]
// ---------------------------------------------------------------------------
__global__ void reduce_a(const float* __restrict__ f0, float* __restrict__ part, int n) {
  __shared__ float s[256];
  int col = threadIdx.x & 31;
  int grp = threadIdx.x >> 5;
  float acc = 0.0f;
  for (int row = blockIdx.x * 8 + grp; row < n; row += gridDim.x * 8)
    acc += f0[(size_t)row * 32 + col];
  s[threadIdx.x] = acc;
  __syncthreads();
  if (threadIdx.x < 32) {
    float t = 0.0f;
    for (int k = 0; k < 8; ++k) t += s[k * 32 + col];
    part[blockIdx.x * 32 + col] = t;
  }
}

__global__ void reduce_b(const float* __restrict__ part, float* __restrict__ out,
                         int nb, float scale) {
  if (threadIdx.x < 32) {
    float t = 0.0f;
    for (int b = 0; b < nb; ++b) t += part[b * 32 + threadIdx.x];
    out[threadIdx.x] = t * scale;
  }
}

// ---------------------------------------------------------------------------
extern "C" void kernel_launch(void* const* d_in, const int* in_sizes, int n_in,
                              void* d_out, int out_size, void* d_ws, size_t ws_size,
                              hipStream_t stream) {
  const float* radii   = (const float*)d_in[0];
  const int*   nbr     = (const int*)d_in[1];
  const int*   charges = (const int*)d_in[2];
  const int N = in_sizes[2];                     // 10000

  const float *w1[3], *b1[3], *w2[3], *b2[3], *w3[3];
  for (int l = 0; l < 3; ++l) {
    w1[l] = (const float*)d_in[3 + 5 * l + 0];
    b1[l] = (const float*)d_in[3 + 5 * l + 1];
    w2[l] = (const float*)d_in[3 + 5 * l + 2];
    b2[l] = (const float*)d_in[3 + 5 * l + 3];
    w3[l] = (const float*)d_in[3 + 5 * l + 4];
  }
  const int NOUTs[3] = {32, 832, 768};

  char* base = (char*)d_ws;
  size_t off = 0;
  auto carve = [&](size_t bytes) -> void* {
    void* p = base + off;
    off += (bytes + 255) & ~(size_t)255;
    return p;
  };
  float* f0a  = (float*)carve((size_t)N * 4);
  float* f0b  = (float*)carve((size_t)N * 16 * 4);
  float* f1b  = (float*)carve((size_t)N * 24 * 4);
  float* f0c  = (float*)carve((size_t)N * 16 * 4);
  float* f1c  = (float*)carve((size_t)N * 24 * 4);
  float* f0d  = (float*)carve((size_t)N * 32 * 4);
  float* part = (float*)carve(64 * 32 * 4);
  _Float16 *w1p[3], *w2p[3], *w3p[3];
  for (int l = 0; l < 3; ++l) {
    w1p[l] = (_Float16*)carve((size_t)7 * 512 * 2);
    w2p[l] = (_Float16*)carve((size_t)28 * 512 * 2);
    w3p[l] = (_Float16*)carve((size_t)(NOUTs[l] / 16) * 4 * 512 * 2);
  }

  for (int l = 0; l < 3; ++l) {
    pack_w<<<16, 256, 0, stream>>>(w1[l], w1p[l], NB, HDIM, 7, 1);
    pack_w<<<64, 256, 0, stream>>>(w2[l], w2p[l], HDIM, HDIM, 7, 4);
    pack_w<<<256, 256, 0, stream>>>(w3[l], w3p[l], HDIM, NOUTs[l], NOUTs[l] / 16, 4);
  }
  prep_f0<<<(N + 255) / 256, 256, 0, stream>>>(charges, f0a, N);

  // 2 waves/block * 2 nodes/wave = 4 nodes per block
  layer_kernel<1, 0, 16, 8, 8, 32><<<N / 4, 64, 0, stream>>>(
      radii, nbr, f0a, nullptr, w1p[0], w2p[0], w3p[0], b1[0], b2[0], f0b, f1b);
  layer_kernel<16, 8, 16, 8, 8, 832><<<N / 4, 64, 0, stream>>>(
      radii, nbr, f0b, f1b, w1p[1], w2p[1], w3p[1], b1[1], b2[1], f0c, f1c);
  layer_kernel<16, 8, 32, 0, 0, 768><<<N / 4, 64, 0, stream>>>(
      radii, nbr, f0c, f1c, w1p[2], w2p[2], w3p[2], b1[2], b2[2], f0d, nullptr);

  reduce_a<<<64, 256, 0, stream>>>(f0d, part, N);
  reduce_b<<<1, 64, 0, stream>>>(part, (float*)d_out, 64, 1.0f / (float)N);

  (void)n_in; (void)out_size; (void)ws_size;
}
